// BaselineGNN_34024730919242
// MI455X (gfx1250) — matrix operand order repaired and checked by reading, not compiled, not run on hardware
//
#include <hip/hip_runtime.h>
#include <hip/hip_bf16.h>

// ---------------------------------------------------------------------------
// GCN forward for MI455X (gfx1250, wave32, WMMA).
//  - Node GEMMs: v_wmma_f32_16x16x32_f16, operands staged in LDS in fragment
//    order (f32->f16 conversion done once at staging, not in the hot loop).
//    B fragments hoisted into registers (4 per wave, reused over 4 row tiles).
//    Branchless fast-path epilogue for full 64-row tiles.
//  - Edge gather/scatter: one wave32 per edge, float4 per lane, f32 atomics.
//    Node arrays (51 MB) are L2-resident (192 MB L2), so the random edge
//    traffic stays on-chip; HBM only sees streaming passes.
// ---------------------------------------------------------------------------

typedef __attribute__((ext_vector_type(16))) _Float16 v16h;
typedef __attribute__((ext_vector_type(8)))  float    v8f;

#define HDIM 128
#define ROWS_PER_BLOCK 64

__device__ __forceinline__ unsigned int pack2h(float lo, float hi) {
    union { _Float16 h[2]; unsigned int u; } t;
    t.h[0] = (_Float16)lo;
    t.h[1] = (_Float16)hi;
    return t.u;
}

// out[nrows x 128] = act( A[nrows x 128] @ W[128 x 128] + bias )
__global__ __launch_bounds__(256)
void gemm128_wmma(const float* __restrict__ A, const float* __restrict__ W,
                  const float* __restrict__ bias, float* __restrict__ out,
                  int nrows, int do_relu) {
    // Fragment-ordered f16 operand staging:
    //  sWfU[((kc*8 + wave)*32 + lane)*8 + j]  : B fragment pairs (32 KB)
    //  sAfU[((kc*4 + rt  )*32 + lane)*8 + j]  : A fragment pairs (16 KB)
    __shared__ __align__(32) unsigned int sWfU[4 * 8 * 32 * 8];
    __shared__ __align__(32) unsigned int sAfU[4 * 4 * 32 * 8];

    const int tid = threadIdx.x;
    const int rowBase = blockIdx.x * ROWS_PER_BLOCK;

    // --- stage W into B-fragment order (8192 f16-pairs / 256 thr = 32 each)
    // B layout (contiguous k order used here, same as A): lane = {half, n}, VGPR j holds K = 16*half + 2j.
    #pragma unroll
    for (int i = 0; i < 32; ++i) {
        int p    = i * 256 + tid;
        int j    = p & 7;
        int lane = (p >> 3) & 31;
        int wv   = (p >> 8) & 7;
        int kc   = p >> 11;
        int half = lane >> 4;
        int c    = wv * 16 + (lane & 15);
        int k    = kc * 32 + 16 * half + 2 * j;
        sWfU[p] = pack2h(W[k * HDIM + c], W[(k + 1) * HDIM + c]);
    }

    // --- stage A tile into A-fragment order (4096 pairs / 256 thr = 16 each)
    // A layout: lane = {half, m}; element e: K = 16*half + e.
    #pragma unroll
    for (int i = 0; i < 16; ++i) {
        int p   = i * 256 + tid;
        int k   = (p & 63) * 2;          // even K; consecutive tids coalesce
        int row = p >> 6;
        float2 v = make_float2(0.0f, 0.0f);
        if (rowBase + row < nrows)
            v = *(const float2*)(A + (size_t)(rowBase + row) * HDIM + k);
        int klo  = k & 31;
        int kc   = k >> 5;
        int half = klo >> 4;
        int e    = klo & 15;                         // even
        int rt   = row >> 4;
        int m    = row & 15;
        sAfU[((kc * 4 + rt) * 32 + half * 16 + m) * 8 + (e >> 1)] = pack2h(v.x, v.y);
    }
    __syncthreads();

    const int wave = tid >> 5;       // 8 waves: one 16-col tile each
    const int lane = tid & 31;
    const int half = lane >> 4;
    const int ln   = lane & 15;
    const int cn   = wave * 16;
    const int col  = cn + ln;

    const v16h* Wf = (const v16h*)sWfU;
    const v16h* Af = (const v16h*)sAfU;

    // Hoist this wave's 4 B fragments into registers (reused by 4 row tiles).
    v16h bf[4];
    #pragma unroll
    for (int kc = 0; kc < 4; ++kc)
        bf[kc] = Wf[(kc * 8 + wave) * 32 + lane];

    // Bias is lane-invariant across the epilogue: load once, branchless after.
    const float bcol = bias ? bias[col] : 0.0f;
    const bool fullTile = (rowBase + ROWS_PER_BLOCK) <= nrows;

    v8f zero = {};
    #pragma unroll
    for (int rt = 0; rt < 4; ++rt) {
        v8f acc = zero;
        #pragma unroll
        for (int kc = 0; kc < 4; ++kc) {
            v16h af = Af[(kc * 4 + rt) * 32 + lane];
            acc = __builtin_amdgcn_wmma_f32_16x16x32_f16(
                false, af, false, bf[kc], (short)0, acc, false, false);
        }
        // Epilogue: C/D layout — VGPR r holds row (r + 8*half), col = lane%16.
        if (fullTile) {
            #pragma unroll
            for (int r = 0; r < 8; ++r) {
                int row = rowBase + rt * 16 + r + 8 * half;
                float v = acc[r] + bcol;
                if (do_relu) v = fmaxf(v, 0.0f);
                out[(size_t)row * HDIM + col] = v;
            }
        } else {
            #pragma unroll
            for (int r = 0; r < 8; ++r) {
                int row = rowBase + rt * 16 + r + 8 * half;
                if (row < nrows) {
                    float v = acc[r] + bcol;
                    if (do_relu) v = fmaxf(v, 0.0f);
                    out[(size_t)row * HDIM + col] = v;
                }
            }
        }
    }
}

// --- degree / normalization ---------------------------------------------
__global__ void deg_init(float* deg, int n) {
    int i = blockIdx.x * blockDim.x + threadIdx.x;
    if (i < n) deg[i] = 1.0f;                    // self loop
}
__global__ void deg_count(const int* __restrict__ dst, float* deg, int e) {
    int i = blockIdx.x * blockDim.x + threadIdx.x;
    if (i < e) atomicAdd(&deg[dst[i]], 1.0f);
}
__global__ void make_dinv(float* d, int n) {
    int i = blockIdx.x * blockDim.x + threadIdx.x;
    if (i < n) d[i] = rsqrtf(d[i]);
}

// --- per-layer elementwise ----------------------------------------------
__global__ void agg_init(const float* __restrict__ hw, const float* __restrict__ dinv,
                         const float* __restrict__ bs, float* __restrict__ agg,
                         long long total) {
    long long i = (long long)blockIdx.x * blockDim.x + threadIdx.x;
    if (i >= total) return;
    int node = (int)(i >> 7), f = (int)(i & 127);
    float di = dinv[node];
    agg[i] = hw[i] * di * di + bs[f];            // self-loop term + layer bias
}

__global__ void bn_relu(const float* __restrict__ agg, const float* __restrict__ gamma,
                        const float* __restrict__ beta, const float* __restrict__ rmean,
                        const float* __restrict__ rvar, float* __restrict__ h,
                        long long total) {
    long long i = (long long)blockIdx.x * blockDim.x + threadIdx.x;
    if (i >= total) return;
    int f = (int)(i & 127);
    float v = (agg[i] - rmean[f]) * rsqrtf(rvar[f] + 1e-5f) * gamma[f] + beta[f];
    h[i] = fmaxf(v, 0.0f);
}

// --- edge scatter: one wave32 per edge, 4 floats per lane ---------------
__global__ __launch_bounds__(256)
void edge_scatter(const int* __restrict__ src, const int* __restrict__ dst,
                  const float* __restrict__ dinv, const float* __restrict__ hw,
                  float* __restrict__ agg, int nedges) {
    int wid  = (blockIdx.x * blockDim.x + threadIdx.x) >> 5;
    int lane = threadIdx.x & 31;
    if (wid >= nedges) return;
    int s = src[wid], d = dst[wid];
    // Warm L2 for the next edge's source row (global_prefetch_b8).
    if (wid + 1 < nedges)
        __builtin_prefetch(hw + (size_t)src[wid + 1] * HDIM + lane * 4, 0, 0);
    float w = dinv[s] * dinv[d];
    float4 v = ((const float4*)(hw + (size_t)s * HDIM))[lane];
    float* dp = agg + (size_t)d * HDIM + lane * 4;
    atomicAdd(dp + 0, v.x * w);
    atomicAdd(dp + 1, v.y * w);
    atomicAdd(dp + 2, v.z * w);
    atomicAdd(dp + 3, v.w * w);
}

// --- pooling -------------------------------------------------------------
__global__ void pool_zero(float* sums, float* cnts, int g) {
    int i = blockIdx.x * blockDim.x + threadIdx.x;
    if (i < g * HDIM) sums[i] = 0.0f;
    if (i < g) cnts[i] = 0.0f;
}
__global__ __launch_bounds__(256)
void pool_acc(const float* __restrict__ h, const int* __restrict__ batch,
              float* __restrict__ sums, float* __restrict__ cnts, int n) {
    int wid  = (blockIdx.x * blockDim.x + threadIdx.x) >> 5;
    int lane = threadIdx.x & 31;
    if (wid >= n) return;
    int g = batch[wid];
    float4 v = ((const float4*)(h + (size_t)wid * HDIM))[lane];
    float* dp = sums + (size_t)g * HDIM + lane * 4;
    atomicAdd(dp + 0, v.x);
    atomicAdd(dp + 1, v.y);
    atomicAdd(dp + 2, v.z);
    atomicAdd(dp + 3, v.w);
    if (lane == 0) atomicAdd(&cnts[g], 1.0f);
}

// --- classifier head: one block per graph -------------------------------
__global__ __launch_bounds__(128)
void head_mlp(const float* __restrict__ sums, const float* __restrict__ cnts,
              const float* __restrict__ fc1w, const float* __restrict__ fc1b,
              const float* __restrict__ fc2w, const float* __restrict__ fc2b,
              float* __restrict__ out) {
    __shared__ float ge[HDIM];
    __shared__ float a1[64];
    int g = blockIdx.x, t = threadIdx.x;
    float inv = 1.0f / fmaxf(cnts[g], 1.0f);
    ge[t] = sums[(size_t)g * HDIM + t] * inv;
    __syncthreads();
    if (t < 64) {
        float acc = fc1b[t];
        #pragma unroll 4
        for (int k = 0; k < HDIM; ++k) acc += ge[k] * fc1w[k * 64 + t];
        a1[t] = fmaxf(acc, 0.0f);
    }
    __syncthreads();
    if (t < 10) {
        float acc = fc2b[t];
        #pragma unroll
        for (int j = 0; j < 64; ++j) acc += a1[j] * fc2w[j * 10 + t];
        out[g * 10 + t] = acc;
    }
}

// ---------------------------------------------------------------------------
extern "C" void kernel_launch(void* const* d_in, const int* in_sizes, int n_in,
                              void* d_out, int out_size, void* d_ws, size_t ws_size,
                              hipStream_t stream) {
    (void)n_in; (void)ws_size;
    const float* x      = (const float*)d_in[0];
    const int*   ei     = (const int*)d_in[1];
    const int*   batch  = (const int*)d_in[2];
    const float* W_in   = (const float*)d_in[3];
    const float* b_in   = (const float*)d_in[4];
    const float* Ws     = (const float*)d_in[5];
    const float* bs     = (const float*)d_in[6];
    const float* gammas = (const float*)d_in[7];
    const float* betas  = (const float*)d_in[8];
    const float* rmeans = (const float*)d_in[9];
    const float* rvars  = (const float*)d_in[10];
    const float* fc1w   = (const float*)d_in[11];
    const float* fc1b   = (const float*)d_in[12];
    const float* fc2w   = (const float*)d_in[13];
    const float* fc2b   = (const float*)d_in[14];
    float* out = (float*)d_out;

    const int N = in_sizes[0] / HDIM;
    const int E = in_sizes[1] / 2;
    const int L = in_sizes[5] / (HDIM * HDIM);
    const int G = out_size / 10;
    const int* src = ei;
    const int* dst = ei + E;

    // Carve scratch (256-B aligned): dinv + h + hw + agg + pool ≈ 155 MB.
    char* p = (char*)d_ws;
    auto carve = [&](size_t bytes) -> void* {
        void* r = (void*)p;
        p += (bytes + 255) & ~(size_t)255;
        return r;
    };
    float* dinv = (float*)carve((size_t)N * 4);
    float* h    = (float*)carve((size_t)N * HDIM * 4);
    float* hw   = (float*)carve((size_t)N * HDIM * 4);
    float* agg  = (float*)carve((size_t)N * HDIM * 4);
    float* sums = (float*)carve((size_t)G * HDIM * 4);
    float* cnts = (float*)carve((size_t)G * 4);

    const dim3 blk(256);
    const long long NF = (long long)N * HDIM;
    const int nGrid    = (N + 255) / 256;
    const int eGrid    = (E + 255) / 256;
    const int nfGrid   = (int)((NF + 255) / 256);
    const int gemmGrid = (N + ROWS_PER_BLOCK - 1) / ROWS_PER_BLOCK;
    const int edgeGrid = (E + 7) / 8;     // 8 wave32 per block, 1 edge per wave
    const int nodeGrid = (N + 7) / 8;

    // Degree -> 1/sqrt(deg)
    deg_init<<<nGrid, blk, 0, stream>>>(dinv, N);
    deg_count<<<eGrid, blk, 0, stream>>>(dst, dinv, E);
    make_dinv<<<nGrid, blk, 0, stream>>>(dinv, N);

    // Input projection: h = relu(x @ W_in + b_in)   [WMMA]
    gemm128_wmma<<<gemmGrid, blk, 0, stream>>>(x, W_in, b_in, h, N, 1);

    for (int l = 0; l < L; ++l) {
        gemm128_wmma<<<gemmGrid, blk, 0, stream>>>(h, Ws + (size_t)l * HDIM * HDIM,
                                                   nullptr, hw, N, 0);
        agg_init<<<nfGrid, blk, 0, stream>>>(hw, dinv, bs + l * HDIM, agg, NF);
        edge_scatter<<<edgeGrid, blk, 0, stream>>>(src, dst, dinv, hw, agg, E);
        bn_relu<<<nfGrid, blk, 0, stream>>>(agg, gammas + l * HDIM, betas + l * HDIM,
                                            rmeans + l * HDIM, rvars + l * HDIM, h, NF);
    }

    pool_zero<<<(G * HDIM + 255) / 256, blk, 0, stream>>>(sums, cnts, G);
    pool_acc<<<nodeGrid, blk, 0, stream>>>(h, batch, sums, cnts, N);
    head_mlp<<<G, 128, 0, stream>>>(sums, cnts, fc1w, fc1b, fc2w, fc2b, out);
}
